// RelationGuidedContrastiveLoss_58162447122710
// MI455X (gfx1250) — matrix operand order repaired and checked by reading, not compiled, not run on hardware
//
#include <hip/hip_runtime.h>
#include <hip/hip_bf16.h>
#include <math.h>

typedef __attribute__((ext_vector_type(16))) _Float16 v16h;
typedef __attribute__((ext_vector_type(8)))  _Float16 v8h;
typedef __attribute__((ext_vector_type(8)))  float    v8f;

#define BATCH 4096
#define DIM   512
#define N2    (2 * BATCH)              // 8192 rows of X
#define CSPLIT 8                       // column groups
#define COLS_PER_GROUP (N2 / CSPLIT)   // 1024
#define TILES_PER_GROUP (COLS_PER_GROUP / 16)  // 64

// ---------------------------------------------------------------- utilities
__device__ __forceinline__ float blockReduceSum256(float v, float* sh) {
#pragma unroll
  for (int m = 16; m > 0; m >>= 1) v += __shfl_xor(v, m, 32);
  const int lane = threadIdx.x & 31;
  const int w    = threadIdx.x >> 5;
  __syncthreads();                 // protect sh reuse across successive calls
  if (lane == 0) sh[w] = v;
  __syncthreads();
  float r = sh[0];
#pragma unroll
  for (int k = 1; k < 8; ++k) r += sh[k];
  return r;                        // broadcast to all threads, deterministic
}

// CDNA5 async global->LDS copy (ISA §15.18.3 op 98, tracked by ASYNCcnt).
__device__ __forceinline__ void async_copy_b128(unsigned lds_off,
                                                const void* gptr) {
  asm volatile("global_load_async_to_lds_b128 %0, %1, off"
               :: "v"(lds_off), "v"(gptr)
               : "memory");
}
__device__ __forceinline__ void wait_asynccnt0() {
  asm volatile("s_wait_asynccnt 0x0" ::: "memory");
}

// ------------------------------------------------- kernel 1: normalize/prep
// grid = BATCH, block = 256.  Writes f16-normalized rows i and i+BATCH of X,
// per-row 1/temperature and pos_sim (already divided by temp).
__global__ void prep_kernel(const float* __restrict__ emb1,
                            const float* __restrict__ emb2,
                            const float* __restrict__ att,
                            _Float16* __restrict__ Xh,
                            float* __restrict__ invT,
                            float* __restrict__ pos) {
  __shared__ float sh[8];
  const int i = blockIdx.x;
  const int t = threadIdx.x;
  const float* p1 = emb1 + (size_t)i * DIM;
  const float* p2 = emb2 + (size_t)i * DIM;
  const float a0 = p1[2 * t], a1 = p1[2 * t + 1];
  const float b0 = p2[2 * t], b1 = p2[2 * t + 1];

  const float s1 = blockReduceSum256(a0 * a0 + a1 * a1, sh);
  const float s2 = blockReduceSum256(b0 * b0 + b1 * b1, sh);
  const float dt = blockReduceSum256(a0 * b0 + a1 * b1, sh);
  const float am = blockReduceSum256(att[(size_t)i * 256 + t], sh);

  const float inv1 = 1.0f / fmaxf(sqrtf(s1), 1e-12f);
  const float inv2 = 1.0f / fmaxf(sqrtf(s2), 1e-12f);
  const float temp = 0.07f * (1.0f + 0.5f * (am * (1.0f / 256.0f)));
  const float it   = 1.0f / temp;

  _Float16* r1 = Xh + (size_t)i * DIM;
  _Float16* r2 = Xh + (size_t)(i + BATCH) * DIM;
  r1[2 * t]     = (_Float16)(a0 * inv1);
  r1[2 * t + 1] = (_Float16)(a1 * inv1);
  r2[2 * t]     = (_Float16)(b0 * inv2);
  r2[2 * t + 1] = (_Float16)(b1 * inv2);

  if (t == 0) {
    invT[i] = it;  invT[i + BATCH] = it;
    const float p = dt * inv1 * inv2 * it;
    pos[i] = p;    pos[i + BATCH] = p;
  }
}

// ---------------------------------- kernel 2: fused WMMA sim -> exp -> rowsum
// grid = (N2/128, CSPLIT), block = 256 (8 waves). Wave w owns 16 rows; the
// workgroup streams 16-column tiles of X through double-buffered LDS using
// async global->LDS copies overlapped with the WMMA chain.
__global__ void __launch_bounds__(256, 1)
simexp_kernel(const _Float16* __restrict__ Xh,
              const float* __restrict__ invT,
              float* __restrict__ partial) {
  __shared__ __align__(16) _Float16 tileB[2][16 * DIM];  // 2 x 16 KiB tiles

  const int lane  = threadIdx.x & 31;
  const int wave  = threadIdx.x >> 5;
  const int khalf = lane >> 4;       // half-wave select (ISA A/B layouts)
  const int l15   = lane & 15;
  const int rowBase      = blockIdx.x * 128 + wave * 16;
  const int colGroup     = blockIdx.y;
  const int colGroupBase = colGroup * COLS_PER_GROUP;

  // stage helper: 16 KiB tile = 1024 16B units, 4 async b128 per thread
  auto stage = [&](int jt, int buf) {
    const int colBase = colGroupBase + jt * 16;
    const char* src = (const char*)(Xh + (size_t)colBase * DIM);
#pragma unroll
    for (int c = 0; c < 4; ++c) {
      const int u = threadIdx.x + 256 * c;               // 16B unit index
      const unsigned lds_off = (unsigned)(size_t)&tileB[buf][u * 8];
      async_copy_b128(lds_off, src + (size_t)u * 16);
    }
  };

  // kick off tile 0 before touching anything else (overlaps A-slab loads)
  stage(0, 0);

  // --- A operand resident in VGPRs: 16 chunks of 16x32 f16 (K = 512) ------
  // ISA 16-bit A 16x32: lanes 0-15 row M, elems 0..7 -> K0..7, 8..15 -> K16..23;
  // lanes 16-31 same rows, K8..15 / K24..31.
  v16h a[16];
  {
    const _Float16* arow = Xh + (size_t)(rowBase + l15) * DIM;
#pragma unroll
    for (int kc = 0; kc < 16; ++kc) {
      const int base = kc * 32 + khalf * 8;
      const v8h lo = *(const v8h*)(arow + base);
      const v8h hi = *(const v8h*)(arow + base + 16);
      a[kc] = __builtin_shufflevector(lo, hi, 0, 1, 2, 3, 4, 5, 6, 7,
                                              8, 9, 10, 11, 12, 13, 14, 15);
    }
  }

  // Per-lane row metadata for the 8 C rows this half-wave sees (M = r + 8*khalf)
  float invT8[8], diagc[8], acc[8];
#pragma unroll
  for (int r = 0; r < 8; ++r) {
    const int row = rowBase + khalf * 8 + r;
    const float it = invT[row];
    invT8[r] = it;
    // diagonal exp(sim_ii) ~ exp(1/temp_i); subtract only in owning col-group
    diagc[r] = ((row >> 10) == colGroup) ? __expf(it) : 0.0f;
    acc[r]   = 0.0f;
  }

  wait_asynccnt0();
  __syncthreads();                   // tile 0 visible to the whole workgroup

  for (int jt = 0; jt < TILES_PER_GROUP; ++jt) {
    const int cur = jt & 1;
    if (jt + 1 < TILES_PER_GROUP)    // async-fetch next tile into other buffer
      stage(jt + 1, (jt + 1) & 1);

    v8f cacc = {};
#pragma unroll
    for (int kc = 0; kc < 16; ++kc) {
      // ISA 16-bit B (KxN): lane n holds column n; half-wave 0 -> K 0..7, 16..23 of
      // chunk, half-wave 1 -> K 8..15, 24..31 (same order as A).
      const _Float16* bp = &tileB[cur][l15 * DIM + kc * 32 + khalf * 8];
      const v8h lo = *(const v8h*)bp;
      const v8h hi = *(const v8h*)(bp + 16);
      const v16h b = __builtin_shufflevector(lo, hi, 0, 1, 2, 3, 4, 5, 6, 7,
                                                     8, 9, 10, 11, 12, 13, 14, 15);
      cacc = __builtin_amdgcn_wmma_f32_16x16x32_f16(
          /*neg_a=*/false, a[kc], /*neg_b=*/false, b,
          /*c_mod=*/(short)0, cacc, /*reuse_a=*/false, /*reuse_b=*/false);
    }

#pragma unroll
    for (int r = 0; r < 8; ++r)
      acc[r] += __expf(cacc[r] * invT8[r]);

    if (jt + 1 < TILES_PER_GROUP)
      wait_asynccnt0();              // my async copies for tile jt+1 done
    __syncthreads();                 // all waves done reading/staging
  }

  // reduce row sums over N: 16 lanes per half-wave (xor masks stay in-half)
#pragma unroll
  for (int r = 0; r < 8; ++r) {
#pragma unroll
    for (int m = 1; m < 16; m <<= 1)
      acc[r] += __shfl_xor(acc[r], m, 32);
  }
  if (l15 == 0) {
#pragma unroll
    for (int r = 0; r < 8; ++r) {
      const int row = rowBase + khalf * 8 + r;
      partial[(size_t)row * CSPLIT + colGroup] = acc[r] - diagc[r];
    }
  }
}

// ------------------------------------------------- kernel 3: final reduction
__global__ void finalize_kernel(const float* __restrict__ partial,
                                const float* __restrict__ pos,
                                float* __restrict__ out) {
  __shared__ float sh[8];
  float local = 0.0f;
  for (int i = threadIdx.x; i < N2; i += 256) {
    float d = 0.0f;
#pragma unroll
    for (int g = 0; g < CSPLIT; ++g) d += partial[(size_t)i * CSPLIT + g];
    local += logf(d) - pos[i];
  }
  local = blockReduceSum256(local, sh);
  if (threadIdx.x == 0) out[0] = local * (1.0f / (float)N2);
}

// --------------------------------------------------------------- launcher
extern "C" void kernel_launch(void* const* d_in, const int* in_sizes, int n_in,
                              void* d_out, int out_size, void* d_ws, size_t ws_size,
                              hipStream_t stream) {
  const float* emb1 = (const float*)d_in[0];
  const float* emb2 = (const float*)d_in[1];
  const float* att  = (const float*)d_in[2];
  float* out = (float*)d_out;

  // workspace layout: f16 X (8 MiB) | invT (32 KiB) | pos (32 KiB) | partials
  char* ws = (char*)d_ws;
  _Float16* Xh   = (_Float16*)ws;
  float*    invT = (float*)(ws + (size_t)N2 * DIM * sizeof(_Float16));
  float*    pos  = invT + N2;
  float*    partial = pos + N2;

  prep_kernel<<<BATCH, 256, 0, stream>>>(emb1, emb2, att, Xh, invT, pos);

  dim3 grid(N2 / 128, CSPLIT);
  simexp_kernel<<<grid, 256, 0, stream>>>(Xh, invT, partial);

  finalize_kernel<<<1, 256, 0, stream>>>(partial, pos, out);
}
